// mgn_GRU_75531294867987
// MI455X (gfx1250) — compile-verified, matrix operand order checked by reading, not run on hardware
//
#include <hip/hip_runtime.h>
#include <hip/hip_bf16.h>
#include <math.h>

typedef __attribute__((ext_vector_type(16))) __bf16 v16bf;
typedef __attribute__((ext_vector_type(8)))  float  v8f;

#define IDIM 64
#define DDIM 128
#define HSTRIDE 8192   /* I*D */
#define BT 32          /* batch rows per block (2 row tiles of 16) */

/* LDS layout (bytes) */
#define HBF_OFF   0        /* h tile bf16: 32 rows x 272B (128 bf16 + 16B pad) */
#define RHBF_OFF  8704     /* r*h tile bf16, same layout */
#define HF32_OFF  17408    /* h tile fp32: 32 rows x 132 floats */
#define HF32_ROW  132
#define XL_OFF    34304    /* 32 rows x 2 floats */
#define ML_OFF    34560    /* 32 floats */
#define SMEM_BYTES 34688

__device__ __forceinline__ unsigned short f2bf(float f) {
  unsigned int u = __float_as_uint(f);
  unsigned int r = 0x7FFFu + ((u >> 16) & 1u);   /* round-to-nearest-even */
  return (unsigned short)((u + r) >> 16);
}

union ABFrag { uint4 u[2]; v16bf v; };

__device__ __forceinline__ float fast_sigmoid(float x) {
  return 1.0f / (1.0f + __expf(-x));
}

/* ------------------------------------------------------------------ */
/* Prepack: U_(r,z,h) (I,D,D) fp32 row-major -> bf16 WMMA B-fragments. */
/* Fragment(g,i,nT,kk) = 32 lanes x 16 bf16, 32B contiguous per lane.  */
/* Per-lane packing matches ISA 7.12.2 16-bit B layout:                */
/*   lane<16:  K = kk*32 + {0..7, 16..23};  lane>=16: +8               */
/* ------------------------------------------------------------------ */
__global__ __launch_bounds__(256) void prepack_u(
    const float* __restrict__ Ur, const float* __restrict__ Uz,
    const float* __restrict__ Uh, unsigned short* __restrict__ ws) {
  unsigned tid  = blockIdx.x * 256u + threadIdx.x;
  unsigned lane = tid & 31u;
  unsigned rem  = tid >> 5;
  unsigned kk = rem & 3u;  rem >>= 2;
  unsigned nT = rem & 7u;  rem >>= 3;
  unsigned i  = rem & 63u; rem >>= 6;
  unsigned g  = rem;                         /* 0..2 */
  if (g >= 3u) return;
  const float* U  = (g == 0u) ? Ur : (g == 1u) ? Uz : Uh;
  const float* Ui = U + (size_t)i * (DDIM * DDIM);
  unsigned khalf = (lane >> 4) * 8u;
  unsigned col   = nT * 16u + (lane & 15u);
  unsigned short tmp[16];
#pragma unroll
  for (int e = 0; e < 16; ++e) {
    unsigned K = kk * 32u + khalf + ((e < 8) ? (unsigned)e : (unsigned)(16 + e - 8));
    tmp[e] = f2bf(Ui[K * DDIM + col]);
  }
  size_t fragIdx = (size_t)(g * IDIM + i) * 32u + nT * 4u + kk;
  unsigned short* dst = ws + fragIdx * 512u + lane * 16u;
  *(uint4*)(dst)     = *(const uint4*)(&tmp[0]);
  *(uint4*)(dst + 8) = *(const uint4*)(&tmp[8]);
}

/* ------------------------------------------------------------------ */
/* Main GRU kernel: one block = (32-row batch tile, group i).          */
/* 8 wave32: wave w owns output columns [16w, 16w+16) for BOTH row     */
/* tiles -> every B fragment load feeds two WMMAs.                     */
/* ------------------------------------------------------------------ */
__global__ __launch_bounds__(256) void gru_wmma(
    const float* __restrict__ X,  const float* __restrict__ M,
    const float* __restrict__ h,
    const float* __restrict__ Wr, const float* __restrict__ Wz,
    const float* __restrict__ Wh,
    const float* __restrict__ br, const float* __restrict__ bz,
    const float* __restrict__ bh,
    const unsigned short* __restrict__ upack,
    float* __restrict__ out) {
  __shared__ __align__(16) unsigned char smem[SMEM_BYTES];
  unsigned short* hbf  = (unsigned short*)(smem + HBF_OFF);
  unsigned short* rhbf = (unsigned short*)(smem + RHBF_OFF);
  float* hf32 = (float*)(smem + HF32_OFF);
  float* Xl   = (float*)(smem + XL_OFF);
  float* Ml   = (float*)(smem + ML_OFF);

  const unsigned bTile = blockIdx.x;
  const unsigned i     = blockIdx.y;
  const unsigned t     = threadIdx.x;
  const unsigned lane  = t & 31u, wave = t >> 5;

  /* ---- stage h tile (32 x 128): fp32 + bf16 copies ---- */
#pragma unroll
  for (int rep = 0; rep < 2; ++rep) {
    unsigned idx  = (unsigned)rep * 2048u + t * 8u;
    unsigned row  = idx >> 7, colb = idx & 127u;
    const float* src = h + (size_t)(bTile * BT + row) * HSTRIDE + i * DDIM + colb;
    float4 f0 = *(const float4*)(src);
    float4 f1 = *(const float4*)(src + 4);
    float* dstf = hf32 + row * HF32_ROW + colb;
    *(float4*)dstf       = f0;
    *(float4*)(dstf + 4) = f1;
    unsigned short* dstb = hbf + row * 136u + colb;
    dstb[0] = f2bf(f0.x); dstb[1] = f2bf(f0.y); dstb[2] = f2bf(f0.z); dstb[3] = f2bf(f0.w);
    dstb[4] = f2bf(f1.x); dstb[5] = f2bf(f1.y); dstb[6] = f2bf(f1.z); dstb[7] = f2bf(f1.w);
  }
  if (t < 64u) Xl[t] = X[(size_t)(bTile * BT + (t >> 1)) * (IDIM * 2) + i * 2u + (t & 1u)];
  if (t < 32u) Ml[t] = M[(size_t)(bTile * BT + t) * IDIM + i];
  __syncthreads();

  const unsigned nT      = wave;               /* column tile 0..7 */
  const unsigned col     = nT * 16u + (lane & 15u);
  const unsigned khalfB  = (lane >> 4) * 16u;  /* byte offset of this lane's K-half */
  const unsigned rowOff  = (lane >> 4) * 8u;   /* C-fragment row base within a 16-row tile */
  const unsigned arow    = (lane & 15u);       /* A-fragment row within a 16-row tile */
  const uint4* up = (const uint4*)upack;

  /* ---- phase 1: r_pre = h@U_r, z_pre = h@U_z (2 row tiles) ---- */
  v8f accR0 = {}, accR1 = {}, accZ0 = {}, accZ1 = {};
#pragma unroll
  for (unsigned kk = 0; kk < 4; ++kk) {
    ABFrag a0, a1, bR, bZ;
    const uint4* ap0 = (const uint4*)(smem + HBF_OFF + arow * 272u + kk * 64u + khalfB);
    const uint4* ap1 = (const uint4*)(smem + HBF_OFF + (arow + 16u) * 272u + kk * 64u + khalfB);
    a0.u[0] = ap0[0]; a0.u[1] = ap0[2];
    a1.u[0] = ap1[0]; a1.u[1] = ap1[2];
    const uint4* pr = up + ((size_t)(0 * IDIM + i) * 32u + nT * 4u + kk) * 64u + lane * 2u;
    bR.u[0] = pr[0]; bR.u[1] = pr[1];
    const uint4* pz = up + ((size_t)(1 * IDIM + i) * 32u + nT * 4u + kk) * 64u + lane * 2u;
    bZ.u[0] = pz[0]; bZ.u[1] = pz[1];
    /* pull phase-2 U_h fragment toward L2/L0 while the matrix pipe is busy */
    __builtin_prefetch((const void*)(up + ((size_t)(2 * IDIM + i) * 32u + nT * 4u + kk) * 64u + lane * 2u), 0, 1);
    accR0 = __builtin_amdgcn_wmma_f32_16x16x32_bf16(false, a0.v, false, bR.v, (short)0, accR0, false, false);
    accR1 = __builtin_amdgcn_wmma_f32_16x16x32_bf16(false, a1.v, false, bR.v, (short)0, accR1, false, false);
    accZ0 = __builtin_amdgcn_wmma_f32_16x16x32_bf16(false, a0.v, false, bZ.v, (short)0, accZ0, false, false);
    accZ1 = __builtin_amdgcn_wmma_f32_16x16x32_bf16(false, a1.v, false, bZ.v, (short)0, accZ1, false, false);
  }

  /* ---- gate activations; write r*h back to LDS in A-fragment layout ---- */
  const float wr0 = Wr[i * 256u + col], wr1 = Wr[i * 256u + 128u + col], brv = br[i * 128u + col];
  const float wz0 = Wz[i * 256u + col], wz1 = Wz[i * 256u + 128u + col], bzv = bz[i * 128u + col];
  float zr[16];
#pragma unroll
  for (int rt = 0; rt < 2; ++rt) {
    const v8f* aR = (rt == 0) ? &accR0 : &accR1;
    const v8f* aZ = (rt == 0) ? &accZ0 : &accZ1;
#pragma unroll
    for (int e = 0; e < 8; ++e) {
      unsigned row = (unsigned)rt * 16u + rowOff + (unsigned)e;
      float x0 = Xl[row * 2u], x1 = Xl[row * 2u + 1u];
      float rv = fast_sigmoid((*aR)[e] + x0 * wr0 + x1 * wr1 + brv);
      zr[rt * 8 + e] = fast_sigmoid((*aZ)[e] + x0 * wz0 + x1 * wz1 + bzv);
      float hv = hf32[row * HF32_ROW + col];
      rhbf[row * 136u + col] = f2bf(rv * hv);
    }
  }
  __syncthreads();

  /* ---- phase 2: h_pre = (r*h)@U_h ---- */
  v8f accH0 = {}, accH1 = {};
#pragma unroll
  for (unsigned kk = 0; kk < 4; ++kk) {
    ABFrag a0, a1, bH;
    const uint4* ap0 = (const uint4*)(smem + RHBF_OFF + arow * 272u + kk * 64u + khalfB);
    const uint4* ap1 = (const uint4*)(smem + RHBF_OFF + (arow + 16u) * 272u + kk * 64u + khalfB);
    a0.u[0] = ap0[0]; a0.u[1] = ap0[2];
    a1.u[0] = ap1[0]; a1.u[1] = ap1[2];
    const uint4* ph = up + ((size_t)(2 * IDIM + i) * 32u + nT * 4u + kk) * 64u + lane * 2u;
    bH.u[0] = ph[0]; bH.u[1] = ph[1];
    accH0 = __builtin_amdgcn_wmma_f32_16x16x32_bf16(false, a0.v, false, bH.v, (short)0, accH0, false, false);
    accH1 = __builtin_amdgcn_wmma_f32_16x16x32_bf16(false, a1.v, false, bH.v, (short)0, accH1, false, false);
  }

  /* ---- h_tilde, blend, store ---- */
  const float wh0 = Wh[i * 256u + col], wh1 = Wh[i * 256u + 128u + col], bhv = bh[i * 128u + col];
#pragma unroll
  for (int rt = 0; rt < 2; ++rt) {
    const v8f* aH = (rt == 0) ? &accH0 : &accH1;
#pragma unroll
    for (int e = 0; e < 8; ++e) {
      unsigned row = (unsigned)rt * 16u + rowOff + (unsigned)e;
      float x0 = Xl[row * 2u], x1 = Xl[row * 2u + 1u];
      float ht = tanhf((*aH)[e] + x0 * wh0 + x1 * wh1 + bhv);
      float hv = hf32[row * HF32_ROW + col];
      float mv = Ml[row];
      float z  = zr[rt * 8 + e];
      float hnew = z * hv + (1.0f - z) * ht;
      out[(size_t)(bTile * BT + row) * HSTRIDE + i * DDIM + col] = hv * (1.0f - mv) + hnew * mv;
    }
  }
}

extern "C" void kernel_launch(void* const* d_in, const int* in_sizes, int n_in,
                              void* d_out, int out_size, void* d_ws, size_t ws_size,
                              hipStream_t stream) {
  const float* X  = (const float*)d_in[0];
  const float* M  = (const float*)d_in[1];
  const float* h  = (const float*)d_in[2];
  const float* Wr = (const float*)d_in[3];
  const float* Wz = (const float*)d_in[4];
  const float* Wh = (const float*)d_in[5];
  const float* Ur = (const float*)d_in[6];
  const float* Uz = (const float*)d_in[7];
  const float* Uh = (const float*)d_in[8];
  const float* br = (const float*)d_in[9];
  const float* bz = (const float*)d_in[10];
  const float* bh = (const float*)d_in[11];
  unsigned short* ws = (unsigned short*)d_ws;   /* needs 3*64*8*4*32*32 B = 6 MB */

  /* 3 gates * 64 groups * 8 nTiles * 4 ksteps * 32 lanes = 196608 threads */
  prepack_u<<<dim3(768), dim3(256), 0, stream>>>(Ur, Uz, Uh, ws);
  gru_wmma<<<dim3(4096 / BT, IDIM), dim3(256), 0, stream>>>(
      X, M, h, Wr, Wz, Wh, br, bz, bh, ws, (float*)d_out);
}
